// RGCN_76278619177084
// MI455X (gfx1250) — matrix operand order repaired and checked by reading, not compiled, not run on hardware
//
#include <hip/hip_runtime.h>
#include <hip/hip_bf16.h>

// Problem constants (match reference)
#define NN   10000
#define EE   160000
#define DD   256
#define RR   237
#define LL   6
#define BB   128
#define KK   33
#define D13  3328          // 13*D

#ifndef __has_builtin
#define __has_builtin(x) 0
#endif
#if __has_builtin(__builtin_amdgcn_global_load_async_to_lds_b128) && \
    __has_builtin(__builtin_amdgcn_s_wait_asynccnt)
#define HAVE_ASYNC 1
#else
#define HAVE_ASYNC 0
#endif

typedef __attribute__((ext_vector_type(16))) __bf16 v16bf;
typedef __attribute__((ext_vector_type(8)))  float  v8f;
typedef int v4i __attribute__((ext_vector_type(4)));

union Frag16 { v16bf v; uint4 u[2]; };

// ---- monotone float<->uint encoding for atomic max/min on floats ----
__device__ __forceinline__ unsigned enc_f32(float f) {
  unsigned u = __float_as_uint(f);
  return (u & 0x80000000u) ? ~u : (u | 0x80000000u);
}
__device__ __forceinline__ float dec_f32(unsigned u) {
  unsigned b = (u & 0x80000000u) ? (u & 0x7FFFFFFFu) : ~u;
  return __uint_as_float(b);
}

__device__ __forceinline__ void atomic_add_f32(float* p, float v) {
  __hip_atomic_fetch_add(p, v, __ATOMIC_RELAXED, __HIP_MEMORY_SCOPE_AGENT);
}
__device__ __forceinline__ void atomic_max_u32(unsigned* p, unsigned v) {
  __hip_atomic_fetch_max(p, v, __ATOMIC_RELAXED, __HIP_MEMORY_SCOPE_AGENT);
}
__device__ __forceinline__ void atomic_min_u32(unsigned* p, unsigned v) {
  __hip_atomic_fetch_min(p, v, __ATOMIC_RELAXED, __HIP_MEMORY_SCOPE_AGENT);
}

// 16-byte global->LDS transfer: async CDNA5 path (ASYNCcnt) or VGPR fallback
__device__ __forceinline__ void copy16_g2l(const __bf16* g, __bf16* l) {
#if HAVE_ASYNC
  __builtin_amdgcn_global_load_async_to_lds_b128(
      (__attribute__((address_space(1))) v4i*)g,
      (__attribute__((address_space(3))) v4i*)l, 0, 0);
#else
  *(uint4*)l = *(const uint4*)g;
#endif
}
__device__ __forceinline__ void wait_stage() {
#if HAVE_ASYNC
  __builtin_amdgcn_s_wait_asynccnt(0);
#endif
}

// ---------------------------------------------------------------------------
// 0) one-time init: x_cur = x ; deg = 1 (self loop) ; logsum = 0
__global__ void k_node_init(const float* __restrict__ x, float* __restrict__ x_cur,
                            float* __restrict__ deg, float* __restrict__ logsum) {
  int tid = blockIdx.x * blockDim.x + threadIdx.x;
  if (tid < NN * DD) x_cur[tid] = x[tid];
  if (tid < NN) deg[tid] = 1.0f;
  if (tid == 0) *logsum = 0.0f;
}

// 1) degree accumulation over edges
__global__ void k_deg_edges(const int* __restrict__ dst, float* __restrict__ deg) {
  int e = blockIdx.x * blockDim.x + threadIdx.x;
  if (e < EE) atomic_add_f32(&deg[dst[e]], 1.0f);
}

// 2) sum of log(deg) (for mean normalization of scale)
__global__ void k_log_reduce(const float* __restrict__ deg, float* __restrict__ logsum) {
  int i = blockIdx.x * blockDim.x + threadIdx.x;
  if (i < NN) atomic_add_f32(logsum, logf(deg[i]));
}

// 3) per-node invariant params: (inv_deg, scale, 1/clip(scale,0.01), 0)
__global__ void k_scales(const float* __restrict__ deg, const float* __restrict__ logsum,
                         float4* __restrict__ params) {
  int i = blockIdx.x * blockDim.x + threadIdx.x;
  if (i >= NN) return;
  float d    = deg[i];
  float mean = (*logsum) / (float)NN;
  float s1   = logf(d) / mean;
  float s2   = 1.0f / fmaxf(s1, 0.01f);
  params[i]  = make_float4(1.0f / d, s1, s2, 0.0f);
}

// ---------------------------------------------------------------------------
// 4) per-layer agg-buffer init with the self-loop message (= x itself)
__global__ void k_agg_init(const float* __restrict__ x,
                           float* __restrict__ ssum, float* __restrict__ ssq,
                           unsigned* __restrict__ smx, unsigned* __restrict__ smn) {
  int tid = blockIdx.x * blockDim.x + threadIdx.x;
  if (tid >= NN * DD) return;
  float v = x[tid];
  ssum[tid] = v;
  ssq[tid]  = v * v;
  unsigned e = enc_f32(v);
  smx[tid] = e;
  smn[tid] = e;
}

// 5) edge scatter: msg = x[src]*rel[etype]; atomically fold into sum/sq/max/min
__global__ void k_edge_scatter(const float* __restrict__ x, const float* __restrict__ rel,
                               const int* __restrict__ src, const int* __restrict__ dst,
                               const int* __restrict__ etype,
                               float* __restrict__ ssum, float* __restrict__ ssq,
                               unsigned* __restrict__ smx, unsigned* __restrict__ smn) {
  int tid = blockIdx.x * blockDim.x + threadIdx.x;
  if (tid >= EE * (DD / 4)) return;
  int e = tid >> 6;          // DD/4 = 64 quads per edge
  int q = tid & 63;
  int s = src[e], dn = dst[e], t = etype[e];
  const float4 xv = *(const float4*)&x[(size_t)s * DD + q * 4];
  const float4 rv = *(const float4*)&rel[(size_t)t * DD + q * 4];
  float m[4] = { xv.x * rv.x, xv.y * rv.y, xv.z * rv.z, xv.w * rv.w };
  size_t base = (size_t)dn * DD + q * 4;
#pragma unroll
  for (int j = 0; j < 4; ++j) {
    atomic_add_f32(&ssum[base + j], m[j]);
    atomic_add_f32(&ssq [base + j], m[j] * m[j]);
    unsigned en = enc_f32(m[j]);
    atomic_max_u32(&smx[base + j], en);
    atomic_min_u32(&smn[base + j], en);
  }
}

// 6) pack GEMM input row A[i, 0:13D] in bf16:
//    agg[j*3+k] = feats[j]*scales[k] (interleaved, as in the reference reshape),
//    last D columns = x.
__global__ void k_pack(const float* __restrict__ ssum, const float* __restrict__ ssq,
                       const unsigned* __restrict__ smx, const unsigned* __restrict__ smn,
                       const float4* __restrict__ params, const float* __restrict__ x,
                       __bf16* __restrict__ A) {
  int tid = blockIdx.x * blockDim.x + threadIdx.x;
  if (tid >= NN * DD) return;
  int i = tid >> 8, d = tid & (DD - 1);
  float4 p   = params[i];
  float mean = ssum[tid] * p.x;
  float sqm  = ssq[tid]  * p.x;
  float sd   = sqrtf(fmaxf(sqm - mean * mean, 1e-6f));
  float feats[4] = { mean, dec_f32(smx[tid]), dec_f32(smn[tid]), sd };
  float sc[3]    = { 1.0f, p.y, p.z };
  __bf16* row = A + (size_t)i * D13;
#pragma unroll
  for (int f = 0; f < 4; ++f) {
    int j = f * DD + d;
#pragma unroll
    for (int k = 0; k < 3; ++k) row[j * 3 + k] = (__bf16)(feats[f] * sc[k]);
  }
  row[12 * DD + d] = (__bf16)x[tid];
}

// 7) W (13D x D fp32, K-major) -> transposed bf16 Wt (D x 13D, N-major)
//    so that GEMM B-tiles are contiguous 16B runs in global memory.
__global__ void k_convW_T(const float* __restrict__ Ws, __bf16* __restrict__ Wt) {
  int tid = blockIdx.x * blockDim.x + threadIdx.x;
  if (tid >= D13 * DD) return;
  int k = tid >> 8;         // coalesced read along n
  int n = tid & (DD - 1);
  Wt[(size_t)n * D13 + k] = (__bf16)Ws[tid];
}

// ---------------------------------------------------------------------------
// 8) GEMM: C(N x D) = A(N x 13D, bf16) * W(13D x D, bf16), f32 accumulate.
//    Block: 256 threads = 8 waves; tile 128(M) x 64(N); K-step 32.
//    Double-buffered LDS staging via GLOBAL_LOAD_ASYNC_TO_LDS_B128 (ASYNCcnt)
//    when available; padded LDS rows (40 bf16) to spread bank accesses.
#define SAW 40
#define SBW 40
__global__ __launch_bounds__(256) void k_gemm(const __bf16* __restrict__ A,
                                              const __bf16* __restrict__ Wt,
                                              float* __restrict__ C) {
  __shared__ __bf16 sA[2][128 * SAW];
  __shared__ __bf16 sB[2][64 * SBW];
  int tid  = threadIdx.x;
  int wave = tid >> 5;
  int lane = tid & 31;
  int row0 = blockIdx.x * 128;
  int col0 = blockIdx.y * 64;

  v8f acc[4] = {};

  auto stage = [&](int buf, int k0) {
    // A tile: 128 rows x 32 k  = 512 x 16B
#pragma unroll
    for (int t = 0; t < 2; ++t) {
      int idx = tid + t * 256;
      int r = idx >> 2, u = idx & 3;
      int row = row0 + r; if (row >= NN) row = NN - 1;
      copy16_g2l(&A[(size_t)row * D13 + k0 + u * 8], &sA[buf][r * SAW + u * 8]);
    }
    // B tile: 64 n-rows x 32 k = 256 x 16B (Wt is N-major: contiguous in k)
    {
      int n = tid >> 2, u = tid & 3;
      copy16_g2l(&Wt[(size_t)(col0 + n) * D13 + k0 + u * 8],
                 &sB[buf][n * SBW + u * 8]);
    }
  };

  int buf = 0;
  stage(0, 0);
  wait_stage();
  __syncthreads();

  for (int k0 = 0; k0 < D13; k0 += 32) {
    if (k0 + 32 < D13) stage(buf ^ 1, k0 + 32);   // overlap with compute

    // A fragment: lanes 0-15 -> M=lane, K in {0..7, 16..23};
    //             lanes 16-31 -> M=lane-16, K in {8..15, 24..31}
    int mrow = lane & 15;
    int kb   = (lane < 16) ? 0 : 8;
    Frag16 fa;
    const __bf16* pa = &sA[buf][(wave * 16 + mrow) * SAW];
    fa.u[0] = *(const uint4*)(pa + kb);
    fa.u[1] = *(const uint4*)(pa + kb + 16);

    int kbb = (lane < 16) ? 0 : 8;    // B: lanes0-15 K={0..7,16..23}, lanes16-31 K={8..15,24..31}
#pragma unroll
    for (int sub = 0; sub < 4; ++sub) {
      Frag16 fb;
      const __bf16* pb = &sB[buf][(sub * 16 + (lane & 15)) * SBW + kbb];
      fb.u[0] = *(const uint4*)(pb);
      fb.u[1] = *(const uint4*)(pb + 16);
      acc[sub] = __builtin_amdgcn_wmma_f32_16x16x32_bf16(
          false, fa.v, false, fb.v, (short)0, acc[sub], false, false);
    }

    wait_stage();        // this wave's async transfers for buf^1 have landed
    __syncthreads();     // all waves' transfers visible; safe to flip
    buf ^= 1;
  }

  // C/D layout: VGPR v -> M = v + (lane<16 ? 0 : 8); N = lane & 15
  int n     = lane & 15;
  int mbase = wave * 16 + ((lane < 16) ? 0 : 8);
#pragma unroll
  for (int sub = 0; sub < 4; ++sub)
#pragma unroll
    for (int v = 0; v < 8; ++v) {
      int row = row0 + mbase + v;
      if (row < NN) C[(size_t)row * DD + col0 + sub * 16 + n] = acc[sub][v];
    }
}

// ---------------------------------------------------------------------------
// 9) bias + layernorm + relu + residual
__global__ void k_post(const float* __restrict__ Cm, const float* __restrict__ bvec,
                       const float* __restrict__ g, const float* __restrict__ be,
                       const float* __restrict__ xin, float* __restrict__ xout) {
  int i = blockIdx.x, d = threadIdx.x;
  float h = Cm[(size_t)i * DD + d] + bvec[d];
  __shared__ float red[DD];
  red[d] = h;
  __syncthreads();
  for (int s = DD / 2; s > 0; s >>= 1) {
    if (d < s) red[d] += red[d + s];
    __syncthreads();
  }
  float mu = red[0] / (float)DD;
  __syncthreads();
  float c = h - mu;
  red[d] = c * c;
  __syncthreads();
  for (int s = DD / 2; s > 0; s >>= 1) {
    if (d < s) red[d] += red[d + s];
    __syncthreads();
  }
  float var = red[0] / (float)DD;
  float y = c * rsqrtf(var + 1e-5f) * g[d] + be[d];
  y = fmaxf(y, 0.0f) + xin[(size_t)i * DD + d];
  xout[(size_t)i * DD + d] = y;
}

// ---------------------------------------------------------------------------
// 10) scoring: one wave per (b,k) triple, shuffle reduction (wave32)
__global__ void k_score(const float* __restrict__ x, const float* __restrict__ relemb,
                        const int* __restrict__ batch, float* __restrict__ out) {
  int pair = blockIdx.x * (blockDim.x >> 5) + (threadIdx.x >> 5);
  int lane = threadIdx.x & 31;
  if (pair >= BB * KK) return;
  int si = batch[pair * 3 + 0];
  int ri = batch[pair * 3 + 1];
  int ti = batch[pair * 3 + 2];
  float acc = 0.0f;
  for (int d = lane; d < DD; d += 32)
    acc += x[(size_t)si * DD + d] * relemb[(size_t)ri * DD + d] * x[(size_t)ti * DD + d];
#pragma unroll
  for (int off = 16; off > 0; off >>= 1) acc += __shfl_xor(acc, off, 32);
  if (lane == 0) out[pair] = acc;
}

// ---------------------------------------------------------------------------
extern "C" void kernel_launch(void* const* d_in, const int* in_sizes, int n_in,
                              void* d_out, int out_size, void* d_ws, size_t ws_size,
                              hipStream_t stream) {
  const float* x        = (const float*)d_in[0];
  const float* rel_embs = (const float*)d_in[1];   // L x R x D
  const float* Ws       = (const float*)d_in[2];   // L x 13D x D
  const float* bs       = (const float*)d_in[3];   // L x D
  const float* ln_scale = (const float*)d_in[4];   // L x D
  const float* ln_bias  = (const float*)d_in[5];   // L x D
  const float* relemb   = (const float*)d_in[6];   // R x D
  const int*   ei       = (const int*)d_in[7];     // 2 x E
  const int*   etype    = (const int*)d_in[8];     // E
  const int*   batch    = (const int*)d_in[9];     // B x K x 3
  float*       out      = (float*)d_out;

  const int* src = ei;
  const int* dst = ei + EE;

  // workspace carve-up
  char* p = (char*)d_ws;
  float*    x_a    = (float*)p;            p += (size_t)NN * DD * 4;
  float*    x_b    = (float*)p;            p += (size_t)NN * DD * 4;
  float*    deg    = (float*)p;            p += (size_t)NN * 4;
  float*    logsum = (float*)p;            p += 16;
  float4*   params = (float4*)p;           p += (size_t)NN * 16;
  float*    ssum   = (float*)p;            p += (size_t)NN * DD * 4;
  float*    ssq    = (float*)p;            p += (size_t)NN * DD * 4;
  unsigned* smx    = (unsigned*)p;         p += (size_t)NN * DD * 4;
  unsigned* smn    = (unsigned*)p;         p += (size_t)NN * DD * 4;
  __bf16*   Abuf   = (__bf16*)p;           p += (size_t)NN * D13 * 2;
  __bf16*   Wtbuf  = (__bf16*)p;           p += (size_t)D13 * DD * 2;
  float*    Cbuf   = (float*)p;            p += (size_t)NN * DD * 4;

  const int T = 256;
  const int gND = (NN * DD + T - 1) / T;
  const int gE  = (EE + T - 1) / T;
  const int gN  = (NN + T - 1) / T;
  const int gEq = (EE * (DD / 4) + T - 1) / T;
  const int gWc = (D13 * DD + T - 1) / T;

  // structure-invariant precompute
  k_node_init <<<gND, T, 0, stream>>>(x, x_a, deg, logsum);
  k_deg_edges <<<gE,  T, 0, stream>>>(dst, deg);
  k_log_reduce<<<gN,  T, 0, stream>>>(deg, logsum);
  k_scales    <<<gN,  T, 0, stream>>>(deg, logsum, params);

  float* cur = x_a;
  float* nxt = x_b;
  for (int l = 0; l < LL; ++l) {
    const float* rel = rel_embs + (size_t)l * RR * DD;
    k_agg_init    <<<gND, T, 0, stream>>>(cur, ssum, ssq, smx, smn);
    k_edge_scatter<<<gEq, T, 0, stream>>>(cur, rel, src, dst, etype, ssum, ssq, smx, smn);
    k_pack        <<<gND, T, 0, stream>>>(ssum, ssq, smx, smn, params, cur, Abuf);
    k_convW_T     <<<gWc, T, 0, stream>>>(Ws + (size_t)l * D13 * DD, Wtbuf);
    dim3 grid((NN + 127) / 128, DD / 64);
    k_gemm        <<<grid, T, 0, stream>>>(Abuf, Wtbuf, Cbuf);
    k_post        <<<NN, DD, 0, stream>>>(Cbuf, bs + (size_t)l * DD,
                                          ln_scale + (size_t)l * DD,
                                          ln_bias  + (size_t)l * DD, cur, nxt);
    float* t = cur; cur = nxt; nxt = t;
  }

  int pairs = BB * KK;
  int gS = (pairs + (T / 32) - 1) / (T / 32);
  k_score<<<gS, T, 0, stream>>>(cur, relemb, batch, out);
}
